// CrfRnnLayer_GBY_4475355923014
// MI455X (gfx1250) — matrix operand order repaired and checked by reading, not compiled, not run on hardware
//
#include <hip/hip_runtime.h>
#include <hip/hip_bf16.h>
#include <stdint.h>

#define H_    112
#define W_    112
#define NPIX  (H_ * W_)          // 12544
#define NCLS  21
#define CPAD  32                 // classes padded; col 21 = ones (denominator)
#define SLABS 4
#define KBLK  32
#define NKB   (NPIX / KBLK)      // 392
#define KB_PER_SLAB (NKB / SLABS)// 98

typedef __attribute__((ext_vector_type(8)))  float        v8f;
typedef __attribute__((ext_vector_type(16))) __bf16       v16bf;
typedef __attribute__((ext_vector_type(8)))  __bf16       v8bf;
typedef __attribute__((ext_vector_type(2)))  __bf16       v2bf;
typedef __attribute__((ext_vector_type(4)))  unsigned int u32x4;
typedef __attribute__((ext_vector_type(8)))  int          i32x8;
typedef __attribute__((ext_vector_type(4)))  int          i32x4;

// Packed 2xf32 -> bf16x2: use v_cvt_pk_bf16_f32 when the toolchain exposes it.
__device__ __forceinline__ unsigned int pack_bf16(float lo, float hi) {
#if __has_builtin(__builtin_amdgcn_cvt_pk_bf16_f32)
    union { v2bf v; unsigned int u; } cv;
    cv.v = __builtin_amdgcn_cvt_pk_bf16_f32(lo, hi);
    return cv.u;
#else
    unsigned int bl = __float_as_uint(lo), bh = __float_as_uint(hi);
    bl = bl + 0x7fffu + ((bl >> 16) & 1u);         // round-to-nearest-even
    bh = bh + 0x7fffu + ((bh >> 16) & 1u);
    return (bl >> 16) | (bh & 0xffff0000u);
#endif
}
__device__ __forceinline__ unsigned short f2bf(float f) {
#if __has_builtin(__builtin_amdgcn_cvt_pk_bf16_f32)
    return (unsigned short)(pack_bf16(f, 0.f) & 0xffffu);
#else
    unsigned int b = __float_as_uint(f);
    b = b + 0x7fffu + ((b >> 16) & 1u);
    return (unsigned short)(b >> 16);
#endif
}

// ---------------------------------------------------------------------------
// Feature build: pre-scaled by sqrt(log2 e) so scores feed v_exp_f32 directly.
// feats[n][16]: [0..2]=spatial fx,fy,bias  [4..9]=bilateral f0..f4,bias
// ---------------------------------------------------------------------------
__global__ void crf_feat(const float* __restrict__ rgb, float* __restrict__ feats) {
    const int n = blockIdx.x * blockDim.x + threadIdx.x;
    if (n >= NPIX) return;
    const float x = (float)(n % W_), y = (float)(n / W_);
    const float SL = 1.2011224087864498f;          // sqrt(log2(e))
    const float fsx = x * (SL / 3.0f), fsy = y * (SL / 3.0f);
    const float sbias = -0.5f * (fsx * fsx + fsy * fsy);
    const float b0 = x * (SL / 160.0f), b1 = y * (SL / 160.0f);
    const float b2 = rgb[n * 3 + 0] * (SL / 3.0f);
    const float b3 = rgb[n * 3 + 1] * (SL / 3.0f);
    const float b4 = rgb[n * 3 + 2] * (SL / 3.0f);
    const float bbias = -0.5f * (b0 * b0 + b1 * b1 + b2 * b2 + b3 * b3 + b4 * b4);
    float* f = feats + (size_t)n * 16;
    f[0] = fsx; f[1] = fsy; f[2] = sbias; f[3] = 0.f;
    f[4] = b0;  f[5] = b1;  f[6] = b2;   f[7] = b3;
    f[8] = b4;  f[9] = bbias;
    f[10] = 0.f; f[11] = 0.f; f[12] = 0.f; f[13] = 0.f; f[14] = 0.f; f[15] = 0.f;
}

// V rows 21..31 are constant: row 21 = 1.0 (denominator column), rest zero.
__global__ void crf_vinit(unsigned short* __restrict__ Vg) {
    const int n = blockIdx.x * blockDim.x + threadIdx.x;
    if (n >= NPIX) return;
    Vg[(size_t)21 * NPIX + n] = 0x3f80;
    #pragma unroll
    for (int c = 22; c < CPAD; ++c) Vg[(size_t)c * NPIX + n] = 0;
}

// ---------------------------------------------------------------------------
// TDM: build D# in SGPR groups and issue tensor_load_to_lds
// (amdgpu-toolchain 6-arg form: g0, g1, g2, g3, extra group, cpol).
// ---------------------------------------------------------------------------
__device__ __forceinline__ void tdm_load_2d(unsigned int lds_addr, const void* gptr,
                                            unsigned int td0, unsigned int td1,
                                            unsigned int tile0, unsigned int tile1,
                                            unsigned long long stride0,
                                            unsigned int dsz_code) {
    unsigned long long ga = (unsigned long long)gptr;
    u32x4 g0;
    g0[0] = 1u;                                              // count=1, user mode
    g0[1] = lds_addr;                                        // LDS byte address
    g0[2] = (unsigned int)ga;                                // global_addr[31:0]
    g0[3] = (unsigned int)((ga >> 32) & 0x01ffffffu) | 0x80000000u; // addr[56:32] | type=2
    i32x8 g1;
    g1[0] = (int)(dsz_code << 16);                           // wg_mask=0, data_size
    g1[1] = (int)((td0 & 0xffffu) << 16);                    // tensor_dim0 lo
    g1[2] = (int)(((td0 >> 16) & 0xffffu) | ((td1 & 0xffffu) << 16));
    g1[3] = (int)(((td1 >> 16) & 0xffffu) | (tile0 << 16));  // tile_dim0
    g1[4] = (int)(tile1 & 0xffffu);                          // tile_dim1, tile_dim2=0
    g1[5] = (int)(stride0 & 0xffffffffu);                    // tensor_dim0_stride lo
    g1[6] = (int)((stride0 >> 32) & 0xffffu);                // stride hi, dim1_stride=0
    g1[7] = 0;
    i32x4 z4 = {0, 0, 0, 0};
    i32x8 z8 = {0, 0, 0, 0, 0, 0, 0, 0};
    __builtin_amdgcn_tensor_load_to_lds(g0, g1, z4, z4, z8, 0);
}

// ---------------------------------------------------------------------------
// Fused dual-kernel attention: out_{s,b}[q, c] += exp(score) * V[k, c]
// Wave = 16-query tile; 32-key blocks staged via TDM; bf16 WMMA for PV.
// ---------------------------------------------------------------------------
__global__ __launch_bounds__(256) void crf_attn(
    const float* __restrict__ feats, const unsigned short* __restrict__ Vg,
    float* __restrict__ part_s, float* __restrict__ part_b) {
    extern __shared__ float smem[];   // [2x512 f32 feats][2x512 f32-sized V bufs] = 8KB
    const int tid  = threadIdx.x;
    const int wave = tid >> 5, lane = tid & 31;
    const int hi   = (lane >> 4) & 1;
    const int q0   = (blockIdx.x * 8 + wave) * 16;
    const int m    = q0 + (lane & 15);
    const int slab = blockIdx.y;

    const float4 qa = *(const float4*)(feats + (size_t)m * 16 + 0); // fx,fy,sbias,-
    const float4 qb = *(const float4*)(feats + (size_t)m * 16 + 4); // b0..b3
    const float2 qc = *(const float2*)(feats + (size_t)m * 16 + 8); // b4,bbias

    v8f acc_s0 = {0.f,0.f,0.f,0.f,0.f,0.f,0.f,0.f};
    v8f acc_s1 = acc_s0, acc_b0 = acc_s0, acc_b1 = acc_s0;

    const unsigned int lds_base = (unsigned int)(size_t)(void*)smem;

    auto issue_block = [&](int b, int buf) {
        const int kb = slab * KB_PER_SLAB + b;
        // key features: contiguous 2KB (32 keys x 16 f32)
        tdm_load_2d(lds_base + buf * 2048, feats + (size_t)kb * KBLK * 16,
                    KBLK * 16, 1, KBLK * 16, 1, KBLK * 16, 2 /*4B*/);
        // V tile: 32 classes x 32 keys bf16, row stride NPIX (true 2-D TDM tile)
        tdm_load_2d(lds_base + 4096 + buf * 2048, Vg + (size_t)kb * KBLK,
                    NPIX, CPAD, KBLK, CPAD, NPIX, 1 /*2B*/);
    };

    if (wave == 0) issue_block(0, 0);

    for (int b = 0; b < KB_PER_SLAB; ++b) {
        __syncthreads();                       // buf[(b+1)&1] free for overwrite
        if (wave == 0) {
            if (b + 1 < KB_PER_SLAB) {
                issue_block(b + 1, (b + 1) & 1);
                __builtin_amdgcn_s_wait_tensorcnt(2);   // block b's 2 loads done
            } else {
                __builtin_amdgcn_s_wait_tensorcnt(0);
            }
        }
        __syncthreads();                       // buf[b&1] published
        const int buf = b & 1;
        const float* kf = smem + buf * 512;
        const char*  vt = (const char*)(smem + 1024 + buf * 512);

        // Scores directly in bf16 A-fragment layout (16-bit A 16x32):
        // reg v, lanes 0-15: K = {2v,2v+1} (v<4) / {2v+8,2v+9} (v>=4); +8 for lanes 16-31
        union { unsigned int u[8]; v16bf v; } As, Ab;
        #pragma unroll
        for (int v = 0; v < 8; ++v) {
            const int k0 = ((v < 4) ? 2 * v : 2 * v + 8) + (hi ? 8 : 0);
            const float* f0 = kf + k0 * 16;
            const float* f1 = f0 + 16;
            const float4 a0 = *(const float4*)f0,       a1 = *(const float4*)f1;
            const float4 g0 = *(const float4*)(f0 + 4), g1 = *(const float4*)(f1 + 4);
            const float2 h0 = *(const float2*)(f0 + 8), h1 = *(const float2*)(f1 + 8);
            const float ss0 = fmaf(qa.x, a0.x, fmaf(qa.y, a0.y, qa.z + a0.z));
            const float ss1 = fmaf(qa.x, a1.x, fmaf(qa.y, a1.y, qa.z + a1.z));
            const float sb0 = fmaf(qb.x, g0.x, fmaf(qb.y, g0.y, fmaf(qb.z, g0.z,
                              fmaf(qb.w, g0.w, fmaf(qc.x, h0.x, qc.y + h0.y)))));
            const float sb1 = fmaf(qb.x, g1.x, fmaf(qb.y, g1.y, fmaf(qb.z, g1.z,
                              fmaf(qb.w, g1.w, fmaf(qc.x, h1.x, qc.y + h1.y)))));
            As.u[v] = pack_bf16(__builtin_amdgcn_exp2f(ss0), __builtin_amdgcn_exp2f(ss1));
            Ab.u[v] = pack_bf16(__builtin_amdgcn_exp2f(sb0), __builtin_amdgcn_exp2f(sb1));
        }

        // B fragments (16-bit B 32x16): lane = class col, K 0-7,16-23 (lo half) / 8-15,24-31 (hi)
        #pragma unroll
        for (int t = 0; t < 2; ++t) {
            const int cl = (lane & 15) + 16 * t;
            const char* bp = vt + cl * 64 + (hi ? 16 : 0);
            const v8bf blo = *(const v8bf*)bp;
            const v8bf bhi = *(const v8bf*)(bp + 32);
            const v16bf Bv = __builtin_shufflevector(blo, bhi,
                0, 1, 2, 3, 4, 5, 6, 7, 8, 9, 10, 11, 12, 13, 14, 15);
            if (t == 0) {
                acc_s0 = __builtin_amdgcn_wmma_f32_16x16x32_bf16(false, As.v, false, Bv,
                                                                 (short)0, acc_s0, false, false);
                acc_b0 = __builtin_amdgcn_wmma_f32_16x16x32_bf16(false, Ab.v, false, Bv,
                                                                 (short)0, acc_b0, false, false);
            } else {
                acc_s1 = __builtin_amdgcn_wmma_f32_16x16x32_bf16(false, As.v, false, Bv,
                                                                 (short)0, acc_s1, false, false);
                acc_b1 = __builtin_amdgcn_wmma_f32_16x16x32_bf16(false, Ab.v, false, Bv,
                                                                 (short)0, acc_b1, false, false);
            }
        }
    }

    // C/D layout: reg r -> row r (lanes 0-15) / r+8 (lanes 16-31); col = lane&15
    float* ps = part_s + ((size_t)slab * NPIX + q0) * 32;
    float* pb = part_b + ((size_t)slab * NPIX + q0) * 32;
    const int col = lane & 15;
    #pragma unroll
    for (int r = 0; r < 8; ++r) {
        const int row = r + hi * 8;
        ps[row * 32 + col]      = acc_s0[r];
        ps[row * 32 + col + 16] = acc_s1[r];
        pb[row * 32 + col]      = acc_b0[r];
        pb[row * 32 + col + 16] = acc_b1[r];
    }
}

// ---------------------------------------------------------------------------
// Per-pixel mixing: combine slabs, normalize, 21x21 matvecs, softmax -> V.
// mode 0: Q=u, softmax->V.  mode 1: full update, softmax->V.  mode 2: write out.
// ---------------------------------------------------------------------------
__global__ __launch_bounds__(256) void crf_mix(
    const float* __restrict__ unaries, const float* __restrict__ SK,
    const float* __restrict__ BK, const float* __restrict__ CM,
    const float* __restrict__ part_s, const float* __restrict__ part_b,
    unsigned short* __restrict__ Vg, float* __restrict__ out, int mode) {
    __shared__ float sk[441], bk[441], cm[441];
    for (int i = threadIdx.x; i < 441; i += 256) { sk[i] = SK[i]; bk[i] = BK[i]; cm[i] = CM[i]; }
    __syncthreads();
    const int n = blockIdx.x * blockDim.x + threadIdx.x;
    if (n >= NPIX) return;

    float q[NCLS];
    if (mode == 0) {
        #pragma unroll
        for (int c = 0; c < NCLS; ++c) q[c] = unaries[(size_t)n * NCLS + c];
    } else {
        float sa[22], ba[22];
        #pragma unroll
        for (int c = 0; c < 22; ++c) { sa[c] = 0.f; ba[c] = 0.f; }
        for (int s = 0; s < SLABS; ++s) {
            const float* ps = part_s + ((size_t)s * NPIX + n) * 32;
            const float* pb = part_b + ((size_t)s * NPIX + n) * 32;
            #pragma unroll
            for (int c = 0; c < 22; ++c) { sa[c] += ps[c]; ba[c] += pb[c]; }
        }
        const float sinv = 1.f / sa[21], binv = 1.f / ba[21];   // denominators (ns, nb)
        #pragma unroll
        for (int c = 0; c < NCLS; ++c) { sa[c] *= sinv; ba[c] *= binv; }
        float w[NCLS];
        #pragma unroll
        for (int co = 0; co < NCLS; ++co) {
            float acc = 0.f;
            #pragma unroll
            for (int c = 0; c < NCLS; ++c)
                acc = fmaf(sk[co * NCLS + c], sa[c], fmaf(bk[co * NCLS + c], ba[c], acc));
            w[co] = acc;
        }
        #pragma unroll
        for (int c = 0; c < NCLS; ++c) {
            float acc = 0.f;
            #pragma unroll
            for (int co = 0; co < NCLS; ++co) acc = fmaf(cm[c * NCLS + co], w[co], acc);
            q[c] = unaries[(size_t)n * NCLS + c] - acc;
        }
    }

    if (mode == 2) {
        #pragma unroll
        for (int c = 0; c < NCLS; ++c) out[(size_t)n * NCLS + c] = q[c];
        return;
    }
    float mx = q[0];
    #pragma unroll
    for (int c = 1; c < NCLS; ++c) mx = fmaxf(mx, q[c]);
    float e[NCLS], ssum = 0.f;
    #pragma unroll
    for (int c = 0; c < NCLS; ++c) {
        e[c] = __builtin_amdgcn_exp2f((q[c] - mx) * 1.44269504f);
        ssum += e[c];
    }
    const float r = 1.f / ssum;
    #pragma unroll
    for (int c = 0; c < NCLS; ++c) Vg[(size_t)c * NPIX + n] = f2bf(e[c] * r);
}

// ---------------------------------------------------------------------------
extern "C" void kernel_launch(void* const* d_in, const int* in_sizes, int n_in,
                              void* d_out, int out_size, void* d_ws, size_t ws_size,
                              hipStream_t stream) {
    (void)in_sizes; (void)n_in; (void)out_size; (void)ws_size;
    const float* unaries = (const float*)d_in[0];
    const float* rgb     = (const float*)d_in[1];
    const float* SK      = (const float*)d_in[2];
    const float* BK      = (const float*)d_in[3];
    const float* CM      = (const float*)d_in[4];
    float* out = (float*)d_out;

    // workspace layout (~14.5 MB total)
    char* ws = (char*)d_ws;
    float*          feats  = (float*)ws;                                   // NPIX*16 f32
    unsigned short* Vg     = (unsigned short*)(ws + (size_t)NPIX * 64);    // CPAD*NPIX bf16
    float*          part_s = (float*)(ws + (size_t)NPIX * 64 + (size_t)CPAD * NPIX * 2);
    float*          part_b = part_s + (size_t)SLABS * NPIX * 32;

    const dim3 blk(256);
    crf_feat <<<NPIX / 256, blk, 0, stream>>>(rgb, feats);
    crf_vinit<<<NPIX / 256, blk, 0, stream>>>(Vg);
    crf_mix  <<<NPIX / 256, blk, 0, stream>>>(unaries, SK, BK, CM, part_s, part_b, Vg, out, 0);
    for (int it = 0; it < 5; ++it) {
        crf_attn<<<dim3(NPIX / 128, SLABS), blk, 8192, stream>>>(feats, Vg, part_s, part_b);
        crf_mix <<<NPIX / 256, blk, 0, stream>>>(unaries, SK, BK, CM, part_s, part_b, Vg, out,
                                                 (it == 4) ? 2 : 1);
    }
}
